// Mamba_25701084299331
// MI455X (gfx1250) — matrix operand order repaired and checked by reading, not compiled, not run on hardware
//
#include <hip/hip_runtime.h>
#include <hip/hip_bf16.h>
#include <stdint.h>

// ---------------------------------------------------------------------------
// Mamba (2 layers) forward for gfx1250.
// GEMMs: v_wmma_f32_16x16x32_bf16, operands pre-converted to bf16 so the
// inner loop is global_load_b128 + wmma only.  One wave owns a 32x64 tile
// (8 accumulators, 8 wmma per K-step).  Tile indices are made wave-uniform
// via readfirstlane so boundary handling is scalar-branched, keeping the hot
// loop free of exec-mask churn.  Selective scan: 1 thread/channel, LDS B/C.
// ---------------------------------------------------------------------------

typedef __attribute__((ext_vector_type(16))) __bf16 v16bf;
typedef __attribute__((ext_vector_type(8)))  float  v8f;
typedef __attribute__((ext_vector_type(4)))  unsigned int v4u;

union bfrag { v16bf v; v4u u[2]; };

#define D_MODEL 1024
#define D_INNER 2048
#define SEQ_L   2048
#define DT_RANK 64
#define D_STATE 16
#define VOCAB   50264
#define XDBL_C  96      // DT_RANK + 2*D_STATE
#define N_LAYER 2

__device__ __forceinline__ unsigned short f2bf(float f) {
  unsigned int u = __builtin_bit_cast(unsigned int, f);
  u += 0x7FFFu + ((u >> 16) & 1u);          // round-to-nearest-even
  return (unsigned short)(u >> 16);
}
__device__ __forceinline__ float siluf(float v) {
  return v / (1.0f + __expf(-v));
}

// ---------------------------------------------------------------------------
// C[M,N] (+)= A[M,K] @ W[N,K]^T,  A/W bf16 row-major, C fp32.
// Wave tile: 32(M) x 64(N).  Fragment layouts per CDNA5 ISA 7.12.2:
//   A (16-bit, 16x32): lane-half h holds row (lane&15), K = {8h..8h+7} ∪ {16+8h..23+8h}
//   B (16-bit, 32x16): lane-half h holds col (lane&15), K = {8h..8h+7} ∪ {16+8h..23+8h}
//   D f32: element r -> M = 8h + r, N = lane&15
// ---------------------------------------------------------------------------
__global__ void __launch_bounds__(256) gemm_bf16_wmma(
    const unsigned short* __restrict__ Ab, int lda,
    const unsigned short* __restrict__ Wb, int ldw,
    float* __restrict__ C, int ldc,
    int M, int N, int K, int addC)
{
  const int lane = threadIdx.x & 31;
  const int wave = (blockIdx.x * blockDim.x + threadIdx.x) >> 5;
  const int nt64 = (N + 63) >> 6;
  // wave-uniform tile coordinates -> SGPRs, scalar control flow
  const int mtile = __builtin_amdgcn_readfirstlane(wave / nt64);
  const int ntile = __builtin_amdgcn_readfirstlane(wave - (wave / nt64) * nt64);
  if (mtile * 32 >= M) return;

  const int half = lane >> 4;
  const int r16  = lane & 15;
  const int m0   = mtile * 32;
  const int ncol = ntile * 64 + r16;

  v8f acc[2][4] = {};

  const unsigned short* aptr0 = Ab + (size_t)(m0 + r16) * lda + 8 * half;
  const unsigned short* aptr1 = aptr0 + (size_t)16 * lda;
  const unsigned short* wptr[4];
#pragma unroll
  for (int t = 0; t < 4; ++t)
    wptr[t] = Wb + (size_t)(ncol + 16 * t) * ldw + 8 * half;

  if (ntile * 64 + 64 <= N) {
    // ---------------- fast path: no bounds checks anywhere ----------------
#pragma unroll 2
    for (int k0 = 0; k0 < K; k0 += 32) {
      bfrag a0, a1, b[4];
      a0.u[0] = *(const v4u*)(aptr0 + k0);
      a0.u[1] = *(const v4u*)(aptr0 + k0 + 16);
      a1.u[0] = *(const v4u*)(aptr1 + k0);
      a1.u[1] = *(const v4u*)(aptr1 + k0 + 16);
#pragma unroll
      for (int t = 0; t < 4; ++t) {
        b[t].u[0] = *(const v4u*)(wptr[t] + k0);
        b[t].u[1] = *(const v4u*)(wptr[t] + k0 + 16);
      }
      __builtin_prefetch(aptr0 + k0 + 64, 0, 1);   // speculative: OOB dropped
      __builtin_prefetch(wptr[0] + k0 + 64, 0, 1);
#pragma unroll
      for (int t = 0; t < 4; ++t) {
        acc[0][t] = __builtin_amdgcn_wmma_f32_16x16x32_bf16(
            false, a0.v, false, b[t].v, (short)0, acc[0][t], false, false);
        acc[1][t] = __builtin_amdgcn_wmma_f32_16x16x32_bf16(
            false, a1.v, false, b[t].v, (short)0, acc[1][t], false, false);
      }
    }
#pragma unroll
    for (int mi = 0; mi < 2; ++mi)
#pragma unroll
      for (int t = 0; t < 4; ++t)
#pragma unroll
        for (int r = 0; r < 8; ++r) {
          size_t ci = (size_t)(m0 + 16 * mi + 8 * half + r) * ldc + ncol + 16 * t;
          float v = acc[mi][t][r];
          if (addC) v += C[ci];
          C[ci] = v;
        }
  } else {
    // ---------------- edge path: last partial N-tile column ----------------
    const v4u vz = {0u, 0u, 0u, 0u};
    for (int k0 = 0; k0 < K; k0 += 32) {
      bfrag a0, a1, b[4];
      a0.u[0] = *(const v4u*)(aptr0 + k0);
      a0.u[1] = *(const v4u*)(aptr0 + k0 + 16);
      a1.u[0] = *(const v4u*)(aptr1 + k0);
      a1.u[1] = *(const v4u*)(aptr1 + k0 + 16);
#pragma unroll
      for (int t = 0; t < 4; ++t) {
        if (ncol + 16 * t < N) {
          b[t].u[0] = *(const v4u*)(wptr[t] + k0);
          b[t].u[1] = *(const v4u*)(wptr[t] + k0 + 16);
        } else {
          b[t].u[0] = vz;
          b[t].u[1] = vz;
        }
      }
#pragma unroll
      for (int t = 0; t < 4; ++t) {
        acc[0][t] = __builtin_amdgcn_wmma_f32_16x16x32_bf16(
            false, a0.v, false, b[t].v, (short)0, acc[0][t], false, false);
        acc[1][t] = __builtin_amdgcn_wmma_f32_16x16x32_bf16(
            false, a1.v, false, b[t].v, (short)0, acc[1][t], false, false);
      }
    }
#pragma unroll
    for (int mi = 0; mi < 2; ++mi)
#pragma unroll
      for (int t = 0; t < 4; ++t) {
        const int n = ncol + 16 * t;
        if (n >= N) continue;
#pragma unroll
        for (int r = 0; r < 8; ++r) {
          size_t ci = (size_t)(m0 + 16 * mi + 8 * half + r) * ldc + n;
          float v = acc[mi][t][r];
          if (addC) v += C[ci];
          C[ci] = v;
        }
      }
  }
}

// ---------------------------------------------------------------------------
// fp32 -> bf16, 8 elements/thread, vectorized b128 in / b128 out.
__global__ void __launch_bounds__(256) cvt_bf16_kernel(
    const float* __restrict__ in, unsigned short* __restrict__ out, int n)
{
  int i = (blockIdx.x * 256 + threadIdx.x) * 8;
  if (i >= n) return;
  float4 x0 = *(const float4*)(in + i);
  float4 x1 = *(const float4*)(in + i + 4);
  v4u o;
  o.x = (unsigned)f2bf(x0.x) | ((unsigned)f2bf(x0.y) << 16);
  o.y = (unsigned)f2bf(x0.z) | ((unsigned)f2bf(x0.w) << 16);
  o.z = (unsigned)f2bf(x1.x) | ((unsigned)f2bf(x1.y) << 16);
  o.w = (unsigned)f2bf(x1.z) | ((unsigned)f2bf(x1.w) << 16);
  *(v4u*)(out + i) = o;
}

__global__ void __launch_bounds__(256) embed_gather(
    const int* __restrict__ ids, const float* __restrict__ emb,
    float* __restrict__ x)
{
  int idx = blockIdx.x * 256 + threadIdx.x;          // over SEQ_L * D_MODEL
  int l = idx >> 10, c = idx & (D_MODEL - 1);
  x[idx] = emb[(size_t)ids[l] * D_MODEL + c];
}

// rmsnorm; output consumed only by GEMMs -> write bf16 directly.
__global__ void __launch_bounds__(256) rmsnorm_bf16_kernel(
    const float* __restrict__ x, const float* __restrict__ w,
    unsigned short* __restrict__ outb)
{
  __shared__ float red[256];
  const int row = blockIdx.x, tid = threadIdx.x;
  const float* xr = x + (size_t)row * D_MODEL;
  float s = 0.f;
  for (int c = tid; c < D_MODEL; c += 256) { float v = xr[c]; s += v * v; }
  red[tid] = s; __syncthreads();
  for (int off = 128; off > 0; off >>= 1) {
    if (tid < off) red[tid] += red[tid + off];
    __syncthreads();
  }
  float rs = rsqrtf(red[0] * (1.0f / D_MODEL) + 1e-5f);
  for (int c = tid; c < D_MODEL; c += 256)
    outb[(size_t)row * D_MODEL + c] = f2bf(xr[c] * rs * w[c]);
}

// causal depthwise conv (k=4, left pad 3) + bias + SiLU.
// Writes fp32 (for scan) and bf16 (for x_proj GEMM).
__global__ void __launch_bounds__(256) conv_silu_kernel(
    const float* __restrict__ xr, const float* __restrict__ cw,
    const float* __restrict__ cb, float* __restrict__ xp,
    unsigned short* __restrict__ xpb)
{
  int idx = blockIdx.x * 256 + threadIdx.x;          // over SEQ_L * D_INNER
  int d = idx & (D_INNER - 1);
  int l = idx >> 11;
  float v = cb[d];
  const float* col = xr + d;
  if (l >= 3) v += cw[d * 4 + 0] * col[(size_t)(l - 3) * (2 * D_INNER)];
  if (l >= 2) v += cw[d * 4 + 1] * col[(size_t)(l - 2) * (2 * D_INNER)];
  if (l >= 1) v += cw[d * 4 + 2] * col[(size_t)(l - 1) * (2 * D_INNER)];
  v += cw[d * 4 + 3] * col[(size_t)l * (2 * D_INNER)];
  float sv = siluf(v);
  xp[idx]  = sv;
  xpb[idx] = f2bf(sv);
}

// selective scan: one thread per channel d; 16-state recurrence in registers.
__global__ void __launch_bounds__(256) scan_kernel(
    const float* __restrict__ xp,    // [L, D_INNER] post conv+silu
    const float* __restrict__ draw,  // [L, D_INNER] dt_proj output (no bias)
    const float* __restrict__ dt_b,  // [D_INNER]
    const float* __restrict__ A_log, // [D_INNER, 16]
    const float* __restrict__ Dp,    // [D_INNER]
    const float* __restrict__ xdbl,  // [L, 96]  cols 64..95 = B,C
    float* __restrict__ y)           // [L, D_INNER]
{
  __shared__ float sbc[64][32];
  const int d = blockIdx.x * 256 + threadIdx.x;
  float a[D_STATE], h[D_STATE];
#pragma unroll
  for (int n = 0; n < D_STATE; ++n) {
    a[n] = -__expf(A_log[(size_t)d * D_STATE + n]);
    h[n] = 0.f;
  }
  const float Dv = Dp[d], db = dt_b[d];

  for (int l0 = 0; l0 < SEQ_L; l0 += 64) {
    __syncthreads();
    for (int j = threadIdx.x; j < 64 * 32; j += 256) {
      int ll = j >> 5, c = j & 31;
      sbc[ll][c] = xdbl[(size_t)(l0 + ll) * XDBL_C + DT_RANK + c];
    }
    __syncthreads();
    for (int ll = 0; ll < 64; ++ll) {
      const int l = l0 + ll;
      float dr = draw[(size_t)l * D_INNER + d] + db;
      float delta = (dr > 20.f) ? dr : log1pf(__expf(dr));
      float u  = xp[(size_t)l * D_INNER + d];
      float du = delta * u;
      float accv = 0.f;
#pragma unroll
      for (int n = 0; n < D_STATE; ++n) {
        float dA = __expf(delta * a[n]);
        h[n] = dA * h[n] + du * sbc[ll][n];
        accv += h[n] * sbc[ll][D_STATE + n];
      }
      y[(size_t)l * D_INNER + d] = accv + u * Dv;
    }
  }
}

// yb = bf16( y * silu(res) ),  res = cols [D_INNER, 2*D_INNER) of xr
__global__ void __launch_bounds__(256) gate_kernel(
    const float* __restrict__ y, const float* __restrict__ xr,
    unsigned short* __restrict__ yb)
{
  int idx = blockIdx.x * 256 + threadIdx.x;          // over SEQ_L * D_INNER
  int d = idx & (D_INNER - 1);
  int l = idx >> 11;
  float r = xr[(size_t)l * (2 * D_INNER) + D_INNER + d];
  yb[idx] = f2bf(y[idx] * siluf(r));
}

// ---------------------------------------------------------------------------
static inline void launch_gemm(const unsigned short* A, int lda,
                               const unsigned short* W, int ldw,
                               float* C, int ldc, int M, int N, int K,
                               int addC, hipStream_t s)
{
  long long mt = (M + 31) / 32, nt = (N + 63) / 64;
  long long waves = mt * nt;
  unsigned blocks = (unsigned)((waves * 32 + 255) / 256);
  gemm_bf16_wmma<<<blocks, 256, 0, s>>>(A, lda, W, ldw, C, ldc, M, N, K, addC);
}

static inline void launch_cvt(const float* in, unsigned short* out, int n,
                              hipStream_t s)
{
  unsigned blocks = (unsigned)((n / 8 + 255) / 256);
  cvt_bf16_kernel<<<blocks, 256, 0, s>>>(in, out, n);
}

extern "C" void kernel_launch(void* const* d_in, const int* in_sizes, int n_in,
                              void* d_out, int out_size, void* d_ws, size_t ws_size,
                              hipStream_t stream) {
  (void)in_sizes; (void)n_in; (void)out_size; (void)ws_size;
  const int*   ids       = (const int*)  d_in[0];
  const float* embedding = (const float*)d_in[1];
  const float* in_proj_w = (const float*)d_in[2];
  const float* conv_w    = (const float*)d_in[3];
  const float* conv_b    = (const float*)d_in[4];
  const float* x_proj_w  = (const float*)d_in[5];
  const float* dt_proj_w = (const float*)d_in[6];
  const float* dt_proj_b = (const float*)d_in[7];
  const float* A_log     = (const float*)d_in[8];
  const float* D_param   = (const float*)d_in[9];
  const float* out_proj_w= (const float*)d_in[10];
  const float* norm_w    = (const float*)d_in[11];
  const float* norm_f_w  = (const float*)d_in[12];
  float* logits = (float*)d_out;

  // ---- workspace carve-out -------------------------------------------------
  // fp32 region
  float* ws   = (float*)d_ws;
  float* x    = ws;                                   // [L, D_MODEL]
  float* xr   = x    + (size_t)SEQ_L * D_MODEL;       // [L, 2*D_INNER]
  float* xp   = xr   + (size_t)SEQ_L * 2 * D_INNER;   // [L, D_INNER]
  float* xdbl = xp   + (size_t)SEQ_L * D_INNER;       // [L, 96]
  float* draw = xdbl + (size_t)SEQ_L * XDBL_C;        // [L, D_INNER]
  float* y    = draw + (size_t)SEQ_L * D_INNER;       // [L, D_INNER]
  // bf16 region (16B-aligned: all sub-sizes are multiples of 8 elements)
  unsigned short* bws   = (unsigned short*)(y + (size_t)SEQ_L * D_INNER);
  unsigned short* xnb   = bws;                                    // [L, D_MODEL]
  unsigned short* xpb   = xnb   + (size_t)SEQ_L * D_MODEL;        // [L, D_INNER]
  unsigned short* xdblb = xpb   + (size_t)SEQ_L * D_INNER;        // [L, 96]
  unsigned short* yb    = xdblb + (size_t)SEQ_L * XDBL_C;         // [L, D_INNER]
  unsigned short* embb  = yb    + (size_t)SEQ_L * D_INNER;        // [VOCAB, D_MODEL]
  unsigned short* ipwb  = embb  + (size_t)VOCAB * D_MODEL;        // [2, 4096, 1024]
  unsigned short* xpwb  = ipwb  + (size_t)N_LAYER * 2 * D_INNER * D_MODEL;
  unsigned short* dtwb  = xpwb  + (size_t)N_LAYER * XDBL_C * D_INNER;
  unsigned short* opwb  = dtwb  + (size_t)N_LAYER * D_INNER * DT_RANK;

  const unsigned eb = (SEQ_L * D_MODEL) / 256;
  const unsigned ib = (SEQ_L * D_INNER) / 256;

  // ---- weight conversion (fp32 -> bf16), once per call ---------------------
  launch_cvt(embedding,  embb, VOCAB * D_MODEL, stream);
  launch_cvt(in_proj_w,  ipwb, N_LAYER * 2 * D_INNER * D_MODEL, stream);
  launch_cvt(x_proj_w,   xpwb, N_LAYER * XDBL_C * D_INNER, stream);
  launch_cvt(dt_proj_w,  dtwb, N_LAYER * D_INNER * DT_RANK, stream);
  launch_cvt(out_proj_w, opwb, N_LAYER * D_MODEL * D_INNER, stream);

  embed_gather<<<eb, 256, 0, stream>>>(ids, embedding, x);

  for (int i = 0; i < N_LAYER; ++i) {
    const unsigned short* ipw = ipwb + (size_t)i * 2 * D_INNER * D_MODEL;
    const unsigned short* xpw = xpwb + (size_t)i * XDBL_C * D_INNER;
    const unsigned short* dtw = dtwb + (size_t)i * D_INNER * DT_RANK;
    const unsigned short* opw = opwb + (size_t)i * D_MODEL * D_INNER;
    const float* cwp = conv_w    + (size_t)i * D_INNER * 4;
    const float* cbp = conv_b    + (size_t)i * D_INNER;
    const float* dtb = dt_proj_b + (size_t)i * D_INNER;
    const float* alg = A_log     + (size_t)i * D_INNER * D_STATE;
    const float* dpp = D_param   + (size_t)i * D_INNER;
    const float* nw  = norm_w    + (size_t)i * D_MODEL;

    rmsnorm_bf16_kernel<<<SEQ_L, 256, 0, stream>>>(x, nw, xnb);
    // in_proj: [L,1024] @ [4096,1024]^T -> [L,4096]
    launch_gemm(xnb, D_MODEL, ipw, D_MODEL, xr, 2 * D_INNER,
                SEQ_L, 2 * D_INNER, D_MODEL, 0, stream);
    conv_silu_kernel<<<ib, 256, 0, stream>>>(xr, cwp, cbp, xp, xpb);
    // x_proj: [L,2048] @ [96,2048]^T -> [L,96]
    launch_gemm(xpb, D_INNER, xpw, D_INNER, xdbl, XDBL_C,
                SEQ_L, XDBL_C, D_INNER, 0, stream);
    launch_cvt(xdbl, xdblb, SEQ_L * XDBL_C, stream);
    // dt_proj: [L,64] (cols 0..63 of xdbl) @ [2048,64]^T -> [L,2048]
    launch_gemm(xdblb, XDBL_C, dtw, DT_RANK, draw, D_INNER,
                SEQ_L, D_INNER, DT_RANK, 0, stream);
    scan_kernel<<<D_INNER / 256, 256, 0, stream>>>(xp, draw, dtb, alg, dpp,
                                                   xdbl, y);
    gate_kernel<<<ib, 256, 0, stream>>>(y, xr, yb);
    // out_proj + residual: x += [L,2048] @ [1024,2048]^T
    launch_gemm(yb, D_INNER, opw, D_INNER, x, D_MODEL,
                SEQ_L, D_MODEL, D_INNER, 1, stream);
  }

  rmsnorm_bf16_kernel<<<SEQ_L, 256, 0, stream>>>(x, norm_f_w, xnb);
  // tied lm_head: [L,1024] @ [VOCAB,1024]^T -> [L,VOCAB]
  launch_gemm(xnb, D_MODEL, embb, D_MODEL, logits, VOCAB,
              SEQ_L, VOCAB, D_MODEL, 0, stream);
}
